// LlamaAttention_77618648973312
// MI455X (gfx1250) — matrix operand order repaired and checked by reading, not compiled, not run on hardware
//
#include <hip/hip_runtime.h>
#include <hip/hip_bf16.h>
#include <stdint.h>

typedef __attribute__((ext_vector_type(16))) _Float16 v16h;
typedef __attribute__((ext_vector_type(8)))  float    v8f;

union HFrag { v16h h; uint32_t u[8]; };
union H2    { _Float16 h[2]; uint32_t u; };

#define S_LEN 2048
#define DIM   4096
#define NHEAD 32
#define NKVH  8
#define HD    128
#define NQ    4096   // NHEAD*HD
#define NKV   1024   // NKVH*HD

// ---- CDNA5 async global->LDS path (guarded; falls back to sync copy) ------
#if defined(__has_builtin)
#  if __has_builtin(__builtin_amdgcn_global_load_async_to_lds_b128) && \
      __has_builtin(__builtin_amdgcn_s_wait_asynccnt)
#    define USE_ASYNC_LDS 1
#  endif
#endif

#ifdef USE_ASYNC_LDS
// The clang builtin's pointer params are in CUDA language address spaces
// (__device__/__shared__), unreachable from HIP casts. Declare the LLVM
// intrinsic directly with target AS(1)/AS(3) params instead (opaque pointers:
// only the AS numbers matter). Param order confirmed by the builtin decl:
// (global src, lds dst, imm offset, imm cpol).
typedef __attribute__((address_space(1))) uint32_t as1_u32;
typedef __attribute__((address_space(3))) uint32_t as3_u32;
extern "C" __device__ void __cdna5_async_load_b128(as1_u32* gsrc, as3_u32* ldst,
                                                   int offset, int cpol)
    __asm("llvm.amdgcn.global.load.async.to.lds.b128");
#endif

// ---------------------------------------------------------------------------
// Generic WMMA GEMM: C[M,N] = A[M,K] * B[K,N]
//   A: fp32 (A_F16=false) or f16 (A_F16=true), row-major
//   B: fp32 row-major, converted to f16 on the fly
//   C: f16 (OUT_F16=true) or fp32, row-major
// Block tile 128x128, K-step 32, 8 waves; register double-buffered global
// loads so WMMA issue overlaps HBM latency.
// ---------------------------------------------------------------------------
struct APack { uint4 q[2]; };            // 16 f16 (one A-row segment)

template<bool A_F16>
__device__ __forceinline__ void load_a_seg(const void* Av, int row, int K,
                                           int kt, int acol, APack& p)
{
    if (A_F16) {
        const _Float16* A = (const _Float16*)Av;
        const uint4* src = (const uint4*)(A + (size_t)row * K + kt * 32 + acol);
        p.q[0] = src[0]; p.q[1] = src[1];
    } else {
        const float* A = (const float*)Av;
        const float4* src = (const float4*)(A + (size_t)row * K + kt * 32 + acol);
        union { _Float16 hh[16]; uint4 q[2]; } tmp;
        #pragma unroll
        for (int i = 0; i < 4; ++i) {
            float4 v = src[i];
            tmp.hh[4*i+0] = (_Float16)v.x;
            tmp.hh[4*i+1] = (_Float16)v.y;
            tmp.hh[4*i+2] = (_Float16)v.z;
            tmp.hh[4*i+3] = (_Float16)v.w;
        }
        p.q[0] = tmp.q[0]; p.q[1] = tmp.q[1];
    }
}

__device__ __forceinline__ void load_b_seg(const float* B, int N, int col0,
                                           int kt, int kp, int n0, uint32_t pk[8])
{
    // two consecutive k-rows (2*kp, 2*kp+1), 8 columns starting at n0;
    // pack (k, k+1) into one dword per column for a single ds_store_b32.
    union F8 { float4 v[2]; float f[8]; };
    F8 r0, r1;
    const float* b0 = B + (size_t)(kt * 32 + 2 * kp) * N + col0 + n0;
    r0.v[0] = ((const float4*)b0)[0];
    r0.v[1] = ((const float4*)b0)[1];
    const float* b1 = b0 + N;
    r1.v[0] = ((const float4*)b1)[0];
    r1.v[1] = ((const float4*)b1)[1];
    #pragma unroll
    for (int i = 0; i < 8; ++i) {
        H2 h2; h2.h[0] = (_Float16)r0.f[i]; h2.h[1] = (_Float16)r1.f[i];
        pk[i] = h2.u;
    }
}

template<bool A_F16, bool OUT_F16>
__global__ __launch_bounds__(256)
void gemm_wmma_kernel(const void* __restrict__ Av, const float* __restrict__ B,
                      void* __restrict__ Cv, int M, int N, int K)
{
    (void)M;
    __shared__ _Float16 Alds[128 * 32];   // [m][k]
    __shared__ _Float16 Blds[128 * 32];   // [n][k]  (transposed)

    const int t    = threadIdx.x;
    const int lane = t & 31;
    const int w    = t >> 5;
    const int wm   = w >> 1;          // 0..3
    const int wn   = w & 1;           // 0..1
    const int mrow = lane & 15;
    const int half = lane >> 4;

    const int row0 = blockIdx.y * 128;
    const int col0 = blockIdx.x * 128;

    v8f acc[2][4] = {};

    // cooperative-load indexing
    const int arow = t >> 1;              // 0..127
    const int acol = (t & 1) << 4;        // 0 / 16
    const int kp   = t >> 4;              // 0..15   (k-pair)
    const int n0   = (t & 15) << 3;       // 0..120  (8 cols)

    const int kTiles = K >> 5;

    // ---- software pipeline: preload tile 0 into registers ----
    APack aP;  uint32_t bPk[8];
    load_a_seg<A_F16>(Av, row0 + arow, K, 0, acol, aP);
    load_b_seg(B, N, col0, 0, kp, n0, bPk);

    for (int kt = 0; kt < kTiles; ++kt) {
        // ---- stage current tile into LDS ----
        {
            uint4* dst = (uint4*)&Alds[arow * 32 + acol];
            dst[0] = aP.q[0]; dst[1] = aP.q[1];
            #pragma unroll
            for (int i = 0; i < 8; ++i)
                *(uint32_t*)&Blds[(n0 + i) * 32 + 2 * kp] = bPk[i];
        }
        __syncthreads();

        // ---- prefetch next tile into registers (overlaps WMMA below) ----
        const int ktn = (kt + 1 < kTiles) ? kt + 1 : kt;
        load_a_seg<A_F16>(Av, row0 + arow, K, ktn, acol, aP);
        load_b_seg(B, N, col0, ktn, kp, n0, bPk);
        if (kt + 2 < kTiles)
            __builtin_prefetch(B + (size_t)((kt + 2) * 32 + 2 * kp) * N + col0 + n0, 0, 1);

        // ---- build all fragments, then back-to-back WMMA ----
        HFrag afr[2];
        #pragma unroll
        for (int mt = 0; mt < 2; ++mt) {
            const int m = wm * 32 + mt * 16 + mrow;
            #pragma unroll
            for (int v = 0; v < 8; ++v) {
                const int k = (v < 4) ? (half * 8 + v * 2) : (16 + half * 8 + (v - 4) * 2);
                afr[mt].u[v] = *(const uint32_t*)&Alds[m * 32 + k];
            }
        }
        HFrag bfr[4];
        #pragma unroll
        for (int nt = 0; nt < 4; ++nt) {
            const int n = wn * 64 + nt * 16 + mrow;
            #pragma unroll
            for (int v = 0; v < 8; ++v)
                bfr[nt].u[v] = *(const uint32_t*)&Blds[n * 32 + ((v < 4) ? (half * 8 + v * 2) : (16 + half * 8 + (v - 4) * 2))];
        }
        #pragma unroll
        for (int nt = 0; nt < 4; ++nt)
            #pragma unroll
            for (int mt = 0; mt < 2; ++mt)
                acc[mt][nt] = __builtin_amdgcn_wmma_f32_16x16x32_f16(
                    false, afr[mt].h, false, bfr[nt].h, (short)0, acc[mt][nt], false, false);
        __syncthreads();
    }

    // ---- epilogue ----
    #pragma unroll
    for (int mt = 0; mt < 2; ++mt) {
        #pragma unroll
        for (int nt = 0; nt < 4; ++nt) {
            const int col   = col0 + wn * 64 + nt * 16 + mrow;
            const int rbase = row0 + wm * 32 + mt * 16 + half * 8;
            #pragma unroll
            for (int r = 0; r < 8; ++r) {
                const float vv = acc[mt][nt][r];
                if (OUT_F16)
                    ((_Float16*)Cv)[(size_t)(rbase + r) * N + col] = (_Float16)vv;
                else
                    ((float*)Cv)[(size_t)(rbase + r) * N + col] = vv;
            }
        }
    }
}

// ---------------------------------------------------------------------------
// RoPE (concatenated-half form), in-place on f16 buffer [S, nheads*128]
// ---------------------------------------------------------------------------
__global__ void rope_kernel(_Float16* __restrict__ buf,
                            const float* __restrict__ cosT,
                            const float* __restrict__ sinT,
                            int nheads)
{
    const int total = S_LEN * nheads * 64;
    const int idx = blockIdx.x * blockDim.x + threadIdx.x;
    if (idx >= total) return;
    const int s   = idx / (nheads * 64);
    const int rem = idx % (nheads * 64);
    const int hh  = rem / 64;
    const int d   = rem % 64;
    const size_t base = (size_t)s * nheads * 128 + hh * 128 + d;
    const float xr = (float)buf[base];
    const float xi = (float)buf[base + 64];
    const float c  = cosT[s * 64 + d];
    const float sn = sinT[s * 64 + d];
    buf[base]      = (_Float16)(xr * c - xi * sn);
    buf[base + 64] = (_Float16)(xr * sn + xi * c);
}

// ---------------------------------------------------------------------------
// Flash attention with GQA (G=4), causal, online softmax.
// grid = (16 q-blocks of 128 rows, 32 heads), 256 threads (8 waves).
// K tile loaded via async global->LDS DMA when available (ASYNCcnt path).
// ---------------------------------------------------------------------------
__global__ __launch_bounds__(256)
void flash_attn_kernel(const _Float16* __restrict__ Q,
                       const _Float16* __restrict__ Kx,
                       const _Float16* __restrict__ V,
                       _Float16* __restrict__ Y)
{
    __shared__ _Float16 Klds [64 * 128];   // [key][dim]
    __shared__ _Float16 Vtlds[128 * 64];   // [dim][key]
    __shared__ _Float16 Plds [8 * 16 * 64];

    const int t    = threadIdx.x;
    const int lane = t & 31;
    const int w    = t >> 5;
    const int qb   = blockIdx.x;
    const int h    = blockIdx.y;
    const int kvh  = h >> 2;              // G = 4
    const int s0   = qb * 128 + w * 16;
    const int mrow = lane & 15;
    const int half = lane >> 4;

    // Q fragments (16 rows x 128 dim -> 4 A-frags), registers for whole kernel
    HFrag qf[4];
    const uint32_t* q32 = (const uint32_t*)Q;
    #pragma unroll
    for (int kc = 0; kc < 4; ++kc) {
        #pragma unroll
        for (int v = 0; v < 8; ++v) {
            const int k = kc * 32 + ((v < 4) ? (half * 8 + v * 2)
                                             : (16 + half * 8 + (v - 4) * 2));
            const size_t idx = (size_t)(s0 + mrow) * NQ + h * 128 + k;
            qf[kc].u[v] = q32[idx >> 1];
        }
    }

    v8f o[8] = {};
    float mst[8], lst[8];
    #pragma unroll
    for (int r = 0; r < 8; ++r) { mst[r] = -1e30f; lst[r] = 0.0f; }

    const int   nkb       = (qb + 1) * 2;            // 64-key blocks (causal)
    const float inv_scale = 0.08838834764831845f;    // 1/sqrt(128)

    const int lkey = t >> 2;            // 0..63  (K-tile copy)
    const int ld0  = (t & 3) * 32;      // 0/32/64/96
    const int vkp  = t >> 3;            // 0..31  (V-tile key pair)
    const int vd0  = (t & 7) * 16;      // 0..112

    for (int jb = 0; jb < nkb; ++jb) {
        __syncthreads();
        const int kb0 = jb * 64;

        // ---- K tile: [key][dim] straight copy (async DMA if available) ----
        {
            const _Float16* gsrc = Kx + (size_t)(kb0 + lkey) * NKV + kvh * 128 + ld0;
            _Float16* ldst = &Klds[lkey * 128 + ld0];
#ifdef USE_ASYNC_LDS
            #pragma unroll
            for (int i = 0; i < 4; ++i)
                __cdna5_async_load_b128((as1_u32*)(void*)(gsrc + i * 8),
                                        (as3_u32*)(void*)(ldst + i * 8), 0, 0);
#else
            const uint4* src = (const uint4*)gsrc;
            uint4* dst = (uint4*)ldst;
            #pragma unroll
            for (int i = 0; i < 4; ++i) dst[i] = src[i];
#endif
        }
        // ---- V tile: transpose into [dim][key], 2 keys packed per dword ----
        {
            union U16 { uint4 q[2]; uint16_t s[16]; };
            U16 va, vb;
            const uint4* sp0 = (const uint4*)(V + (size_t)(kb0 + 2 * vkp) * NKV + kvh * 128 + vd0);
            va.q[0] = sp0[0]; va.q[1] = sp0[1];
            const uint4* sp1 = (const uint4*)(V + (size_t)(kb0 + 2 * vkp + 1) * NKV + kvh * 128 + vd0);
            vb.q[0] = sp1[0]; vb.q[1] = sp1[1];
            #pragma unroll
            for (int i = 0; i < 16; ++i) {
                const uint32_t pk = (uint32_t)va.s[i] | ((uint32_t)vb.s[i] << 16);
                *(uint32_t*)&Vtlds[(vd0 + i) * 64 + 2 * vkp] = pk;
            }
        }
        if (jb + 1 < nkb) {
            __builtin_prefetch(Kx + (size_t)(kb0 + 64 + lkey) * NKV + kvh * 128 + ld0, 0, 1);
            __builtin_prefetch(V  + (size_t)(kb0 + 64 + lkey) * NKV + kvh * 128 + ld0, 0, 1);
        }
#ifdef USE_ASYNC_LDS
        __builtin_amdgcn_s_wait_asynccnt(0);
#endif
        __syncthreads();

        // ---- S = Q @ K^T : 16x64 per wave (batched frags, b2b wmma) ----
        v8f sA[4] = {};
        #pragma unroll
        for (int kc = 0; kc < 4; ++kc) {
            HFrag bf[4];
            #pragma unroll
            for (int nt = 0; nt < 4; ++nt) {
                const int n = nt * 16 + mrow;
                #pragma unroll
                for (int v = 0; v < 8; ++v)
                    bf[nt].u[v] = *(const uint32_t*)&Klds[n * 128 + kc * 32 + ((v < 4) ? (half * 8 + v * 2) : (16 + half * 8 + (v - 4) * 2))];
            }
            #pragma unroll
            for (int nt = 0; nt < 4; ++nt)
                sA[nt] = __builtin_amdgcn_wmma_f32_16x16x32_f16(
                    false, qf[kc].h, false, bf[nt].h, (short)0, sA[nt], false, false);
        }

        // ---- scale + causal mask ----
        #pragma unroll
        for (int nt = 0; nt < 4; ++nt) {
            const int kg = kb0 + nt * 16 + mrow;
            #pragma unroll
            for (int r = 0; r < 8; ++r) {
                const int qg = s0 + r + half * 8;
                const float vv = sA[nt][r] * inv_scale;
                sA[nt][r] = (kg > qg) ? -1e9f : vv;
            }
        }

        // ---- online softmax (rows live on 16-lane halves) ----
        #pragma unroll
        for (int r = 0; r < 8; ++r) {
            float mloc = fmaxf(fmaxf(sA[0][r], sA[1][r]), fmaxf(sA[2][r], sA[3][r]));
            #pragma unroll
            for (int off = 8; off >= 1; off >>= 1)
                mloc = fmaxf(mloc, __shfl_xor(mloc, off, 32));
            const float mnew  = fmaxf(mst[r], mloc);
            const float alpha = __expf(mst[r] - mnew);
            float rsum = 0.0f;
            #pragma unroll
            for (int nt = 0; nt < 4; ++nt) {
                const float p = __expf(sA[nt][r] - mnew);
                sA[nt][r] = p;
                rsum += p;
            }
            #pragma unroll
            for (int off = 8; off >= 1; off >>= 1)
                rsum += __shfl_xor(rsum, off, 32);
            lst[r] = lst[r] * alpha + rsum;
            mst[r] = mnew;
            #pragma unroll
            for (int nt = 0; nt < 8; ++nt) o[nt][r] *= alpha;
        }

        // ---- stage P (C-layout -> LDS -> A-layout) ----
        _Float16* Pw = &Plds[w * 16 * 64];
        #pragma unroll
        for (int nt = 0; nt < 4; ++nt) {
            const int col = nt * 16 + mrow;
            #pragma unroll
            for (int r = 0; r < 8; ++r)
                Pw[(r + half * 8) * 64 + col] = (_Float16)sA[nt][r];
        }

        // ---- O += P @ V ----
        #pragma unroll
        for (int kc = 0; kc < 2; ++kc) {
            HFrag pf;
            #pragma unroll
            for (int v = 0; v < 8; ++v) {
                const int k = kc * 32 + ((v < 4) ? (half * 8 + v * 2)
                                                 : (16 + half * 8 + (v - 4) * 2));
                pf.u[v] = *(const uint32_t*)&Pw[mrow * 64 + k];
            }
            #pragma unroll
            for (int ng = 0; ng < 2; ++ng) {
                HFrag bf[4];
                #pragma unroll
                for (int j = 0; j < 4; ++j) {
                    const int n = (ng * 4 + j) * 16 + mrow;
                    #pragma unroll
                    for (int v = 0; v < 8; ++v)
                        bf[j].u[v] = *(const uint32_t*)&Vtlds[n * 64 + kc * 32 + ((v < 4) ? (half * 8 + v * 2) : (16 + half * 8 + (v - 4) * 2))];
                }
                #pragma unroll
                for (int j = 0; j < 4; ++j)
                    o[ng * 4 + j] = __builtin_amdgcn_wmma_f32_16x16x32_f16(
                        false, pf.h, false, bf[j].h, (short)0, o[ng * 4 + j], false, false);
            }
        }
    }

    // ---- normalize + write Y (f16) ----
    #pragma unroll
    for (int nt = 0; nt < 8; ++nt) {
        const int col = h * 128 + nt * 16 + mrow;
        #pragma unroll
        for (int r = 0; r < 8; ++r) {
            const int row = s0 + r + half * 8;
            Y[(size_t)row * NQ + col] = (_Float16)(o[nt][r] / lst[r]);
        }
    }
}

// ---------------------------------------------------------------------------
extern "C" void kernel_launch(void* const* d_in, const int* in_sizes, int n_in,
                              void* d_out, int out_size, void* d_ws, size_t ws_size,
                              hipStream_t stream) {
    (void)in_sizes; (void)n_in; (void)out_size; (void)ws_size;
    const float* hidden = (const float*)d_in[0];
    const float* cosT   = (const float*)d_in[1];
    const float* sinT   = (const float*)d_in[2];
    // d_in[3] = atten_mask (causal handled analytically)
    const float* Wq     = (const float*)d_in[4];
    const float* Wk     = (const float*)d_in[5];
    const float* Wv     = (const float*)d_in[6];
    const float* Wo     = (const float*)d_in[7];
    float* out = (float*)d_out;

    char* ws = (char*)d_ws;
    _Float16* Qb = (_Float16*)(ws);                            // 16 MB
    _Float16* Kb = (_Float16*)(ws + (size_t)16 * 1024 * 1024); //  4 MB
    _Float16* Vb = (_Float16*)(ws + (size_t)20 * 1024 * 1024); //  4 MB
    _Float16* Yb = (_Float16*)(ws + (size_t)24 * 1024 * 1024); // 16 MB

    dim3 blk(256);
    gemm_wmma_kernel<false, true><<<dim3(NQ / 128,  S_LEN / 128), blk, 0, stream>>>(hidden, Wq, Qb, S_LEN, NQ,  DIM);
    gemm_wmma_kernel<false, true><<<dim3(NKV / 128, S_LEN / 128), blk, 0, stream>>>(hidden, Wk, Kb, S_LEN, NKV, DIM);
    gemm_wmma_kernel<false, true><<<dim3(NKV / 128, S_LEN / 128), blk, 0, stream>>>(hidden, Wv, Vb, S_LEN, NKV, DIM);
    {
        int totq = S_LEN * NHEAD * 64;
        rope_kernel<<<(totq + 255) / 256, blk, 0, stream>>>(Qb, cosT, sinT, NHEAD);
        int totk = S_LEN * NKVH * 64;
        rope_kernel<<<(totk + 255) / 256, blk, 0, stream>>>(Kb, cosT, sinT, NKVH);
    }
    flash_attn_kernel<<<dim3(S_LEN / 128, NHEAD), blk, 0, stream>>>(Qb, Kb, Vb, Yb);
    gemm_wmma_kernel<true, false><<<dim3(DIM / 128, S_LEN / 128), blk, 0, stream>>>(Yb, Wo, out, S_LEN, DIM, NQ);
}
